// RSSM_64596308132506
// MI455X (gfx1250) — matrix operand order repaired and checked by reading, not compiled, not run on hardware
//
#include <hip/hip_runtime.h>
#include <hip/hip_bf16.h>
#include <math.h>

typedef __bf16 bf16;
typedef __attribute__((ext_vector_type(16))) __bf16 v16bf;
typedef __attribute__((ext_vector_type(8)))  __bf16 v8bf;
typedef __attribute__((ext_vector_type(4)))  __bf16 v4bf;
typedef __attribute__((ext_vector_type(8)))  float  v8f;

#define T_STEPS 64
#define BATCH   1024
#define DETER   512
#define STOC    64
#define EMB     1024
#define ACT     6
#define MLP     512
#define OUTW    896   // DETER + 6*STOC
#define KIN     96    // STOC+ACT=70 padded to 3 k-tiles of 32

// f32 LDS strides (floats)
#define SHF_STR 520   // 512 + 8  (8-bank skew)
#define SPR_STR 136
// bf16 LDS strides (elements) — multiples of 8 so 16B-aligned fragment loads
#define SB_STR   528  // 512 + 16 (skew: 1056B/row -> 8-bank rotation)
#define SEMB_STR 1040 // 1024 + 16
#define SIN_STR  112  // 96 + 16

// ---------------------------------------------------------------------------
// WMMA helpers (CDNA5 16x16x32 bf16 -> f32)
// ---------------------------------------------------------------------------
__device__ __forceinline__ v8f wmma_bf16(v16bf a, v16bf b, v8f c) {
  return __builtin_amdgcn_wmma_f32_16x16x32_bf16(false, a, false, b, (short)0, c,
                                                 false, false);
}

__device__ __forceinline__ v16bf cat8(v8bf lo, v8bf hi) {
  return __builtin_shufflevector(lo, hi, 0, 1, 2, 3, 4, 5, 6, 7,
                                 8, 9, 10, 11, 12, 13, 14, 15);
}

// A-matrix 16x32 (MxK) fragment from bf16 LDS (row-major, stride elements).
// Lane l<16: row=l, K groups [k0+0..7] and [k0+16..23]; lanes 16-31: +8.
// Two 16B ds_load_b128, zero VALU.
__device__ __forceinline__ v16bf a_frag_lds(const bf16* s, int stride, int k0) {
  const int lane = threadIdx.x & 31;
  const bf16* p = s + (lane & 15) * stride + k0 + (lane >> 4) * 8;
  return cat8(*(const v8bf*)p, *(const v8bf*)(p + 16));
}

// B-matrix 32x16 (KxN) fragment from bf16 weights stored [N][K], K contiguous.
// Lane l<16: col N=n0+l, K groups [k0+0..7] and [k0+16..23]; lanes 16-31: +8.
__device__ __forceinline__ v16bf b_frag_glb(const bf16* __restrict__ W, int K,
                                            int n0, int k0) {
  const int lane = threadIdx.x & 31;
  const bf16* p = W + (size_t)(n0 + (lane & 15)) * K + k0 + ((lane >> 4) << 3);
  return cat8(*(const v8bf*)p, *(const v8bf*)(p + 16));
}

__device__ __forceinline__ float elu_f(float x) {
  return x > 0.f ? x : (__expf(x) - 1.f);
}
__device__ __forceinline__ float sigmoid_f(float x) {
  return 1.f / (1.f + __expf(-x));
}
__device__ __forceinline__ float softplus_f(float x) {
  return x > 20.f ? x : log1pf(__expf(x));
}

// ---------------------------------------------------------------------------
// Weight preprocessing: fp32 -> bf16 (with transpose / padding where needed)
// ---------------------------------------------------------------------------
__global__ void cvt_kernel(const float* __restrict__ in, bf16* __restrict__ out,
                           int n) {
  int idx = blockIdx.x * blockDim.x + threadIdx.x;
  if (idx < n) out[idx] = (bf16)in[idx];
}

// in[K][N] row-major -> out[N][K] bf16 (K contiguous)
__global__ void transpose_cvt_kernel(const float* __restrict__ in,
                                     bf16* __restrict__ out, int K, int N) {
  int idx = blockIdx.x * blockDim.x + threadIdx.x;
  if (idx >= K * N) return;
  int n = idx / K, k = idx % K;
  out[idx] = (bf16)in[(size_t)k * N + n];
}

// Wi [70][512] -> WiT [512][96] bf16, zero-padded K 70..95
__global__ void wi_pad_kernel(const float* __restrict__ Wi, bf16* __restrict__ out) {
  int idx = blockIdx.x * blockDim.x + threadIdx.x;
  if (idx >= DETER * KIN) return;
  int n = idx / KIN, k = idx % KIN;
  out[idx] = (k < STOC + ACT) ? (bf16)Wi[(size_t)k * DETER + n] : (bf16)0.f;
}

// ---------------------------------------------------------------------------
// One RSSM timestep. Grid = B/16 workgroups; each WG owns 16 batch rows and
// runs the whole step (input MLP, GRU, prior head, posterior head) via WMMA.
// 256 threads = 8 wave32 waves; waves split the N dimension of each GEMM.
// Activations live in LDS as bf16 (converted once at production time).
// ---------------------------------------------------------------------------
__global__ __launch_bounds__(256)
void rssm_step(int t,
               const float* __restrict__ actions,
               const float* __restrict__ nonterm,
               const float* __restrict__ emb,
               const float* __restrict__ init_deter,
               const float* __restrict__ init_stoc,
               const float* __restrict__ noise_p,
               const float* __restrict__ noise_q,
               const float* __restrict__ bi,
               const float* __restrict__ b_ih,
               const float* __restrict__ b_hh,
               const float* __restrict__ bp1,
               const float* __restrict__ bp2,
               const float* __restrict__ bq1,
               const float* __restrict__ bq2,
               const bf16* __restrict__ WiT,
               const bf16* __restrict__ Wih,
               const bf16* __restrict__ Whh,
               const bf16* __restrict__ Wp1T,
               const bf16* __restrict__ Wp2T,
               const bf16* __restrict__ Wq1T,
               const bf16* __restrict__ Wq2T,
               float* __restrict__ out) {
  __shared__ float sHf[16 * SHF_STR];    // h = deter_prev*nt (fp32, gate term)
  __shared__ float sPR[16 * SPR_STR];    // raw head outputs (16 x 128)
  __shared__ bf16 sXb[16 * SB_STR];      // x / prior hidden / post hidden
  __shared__ bf16 sHb[16 * SB_STR];      // h (bf16 GEMM operand)
  __shared__ bf16 sDb[16 * SB_STR];      // deter_new (bf16 GEMM operand)
  __shared__ bf16 sInb[16 * SIN_STR];    // [stoc*nt | act | 0-pad]
  __shared__ bf16 sEmb[16 * SEMB_STR];   // obs embedding tile

  const int tid  = threadIdx.x;
  const int lane = tid & 31;
  const int wave = tid >> 5;
  const int hi   = lane >> 4;
  const int colL = lane & 15;
  const int m0   = blockIdx.x * 16;

  const float* prevOut = (t > 0) ? (out + (size_t)(t - 1) * BATCH * OUTW) : nullptr;
  float* curOut = out + (size_t)t * BATCH * OUTW;

  // ---- Phase A: stage recurrent inputs + emb into LDS ---------------------
  // deter_prev * nt -> sHf (f32) and sHb (bf16); vectorized float4 reads.
  for (int idx = tid; idx < 16 * (DETER / 4); idx += 256) {
    int r = idx >> 7, c4 = (idx & 127) * 4;
    int b = m0 + r;
    float nt = nonterm[(size_t)t * BATCH + b];
    const float* src = prevOut ? (prevOut + (size_t)b * OUTW + c4)
                               : (init_deter + (size_t)b * DETER + c4);
    float4 v = *(const float4*)src;
    v.x *= nt; v.y *= nt; v.z *= nt; v.w *= nt;
    float* pf = sHf + r * SHF_STR + c4;
    pf[0] = v.x; pf[1] = v.y; pf[2] = v.z; pf[3] = v.w;
    v4bf w; w[0] = (bf16)v.x; w[1] = (bf16)v.y; w[2] = (bf16)v.z; w[3] = (bf16)v.w;
    *(v4bf*)(sHb + r * SB_STR + c4) = w;
  }
  // obs embedding tile -> sEmb (bf16), converted exactly once per element.
  for (int idx = tid; idx < 16 * (EMB / 4); idx += 256) {
    int r = idx >> 8, c4 = (idx & 255) * 4;
    float4 v = *(const float4*)(emb + ((size_t)t * BATCH + m0 + r) * EMB + c4);
    v4bf w; w[0] = (bf16)v.x; w[1] = (bf16)v.y; w[2] = (bf16)v.z; w[3] = (bf16)v.w;
    *(v4bf*)(sEmb + r * SEMB_STR + c4) = w;
  }
  // [stoc*nt | actions | zero-pad] -> sInb
  for (int idx = tid; idx < 16 * SIN_STR; idx += 256) {
    int r = idx / SIN_STR, c = idx % SIN_STR;
    int b = m0 + r;
    float v = 0.f;
    if (c < STOC) {
      float nt = nonterm[(size_t)t * BATCH + b];
      float st = prevOut ? prevOut[(size_t)b * OUTW + DETER + 5 * STOC + c]
                         : init_stoc[(size_t)b * STOC + c];
      v = st * nt;
    } else if (c < STOC + ACT) {
      v = actions[((size_t)t * BATCH + b) * ACT + (c - STOC)];
    }
    sInb[r * SIN_STR + c] = (bf16)v;
  }
  __syncthreads();

  // ---- Phase B: x = elu(sIn @ WiT + bi) -> sXb  (M16,K96,N512) ------------
  for (int j = wave; j < DETER / 16; j += 8) {
    int n0 = j * 16;
    v8f acc = {};
#pragma unroll
    for (int k0 = 0; k0 < KIN; k0 += 32)
      acc = wmma_bf16(a_frag_lds(sInb, SIN_STR, k0),
                      b_frag_glb(WiT, KIN, n0, k0), acc);
    int col = n0 + colL;
    float bias = bi[col];
#pragma unroll
    for (int v = 0; v < 8; ++v)
      sXb[(v + hi * 8) * SB_STR + col] = (bf16)elu_f(acc[v] + bias);
  }
  __syncthreads();

  // ---- Phase C: GRU — six fused GEMM accumulations + gates ----------------
  for (int j = wave; j < DETER / 16; j += 8) {
    int n0 = j * 16;
    v8f aIR = {}, aIZ = {}, aIN = {}, aHR = {}, aHZ = {}, aHN = {};
#pragma unroll 2
    for (int k0 = 0; k0 < DETER; k0 += 32) {
      v16bf ax = a_frag_lds(sXb, SB_STR, k0);
      v16bf ah = a_frag_lds(sHb, SB_STR, k0);
      aIR = wmma_bf16(ax, b_frag_glb(Wih, DETER, n0,             k0), aIR);
      aIZ = wmma_bf16(ax, b_frag_glb(Wih, DETER, n0 + DETER,     k0), aIZ);
      aIN = wmma_bf16(ax, b_frag_glb(Wih, DETER, n0 + 2 * DETER, k0), aIN);
      aHR = wmma_bf16(ah, b_frag_glb(Whh, DETER, n0,             k0), aHR);
      aHZ = wmma_bf16(ah, b_frag_glb(Whh, DETER, n0 + DETER,     k0), aHZ);
      aHN = wmma_bf16(ah, b_frag_glb(Whh, DETER, n0 + 2 * DETER, k0), aHN);
    }
    int col = n0 + colL;
    float bir = b_ih[col], biz = b_ih[col + DETER], bin = b_ih[col + 2 * DETER];
    float bhr = b_hh[col], bhz = b_hh[col + DETER], bhn = b_hh[col + 2 * DETER];
#pragma unroll
    for (int v = 0; v < 8; ++v) {
      int row = v + hi * 8;
      float r  = sigmoid_f(aIR[v] + bir + aHR[v] + bhr);
      float z  = sigmoid_f(aIZ[v] + biz + aHZ[v] + bhz);
      float nn = tanhf(aIN[v] + bin + r * (aHN[v] + bhn));
      float h  = sHf[row * SHF_STR + col];
      float d  = (1.f - z) * nn + z * h;
      sDb[row * SB_STR + col] = (bf16)d;
      curOut[(size_t)(m0 + row) * OUTW + col] = d;
    }
  }
  __syncthreads();

  // ---- Phase D: prior hidden = elu(sD @ Wp1T + bp1) -> sXb ----------------
  for (int j = wave; j < MLP / 16; j += 8) {
    int n0 = j * 16;
    v8f acc = {};
#pragma unroll 4
    for (int k0 = 0; k0 < DETER; k0 += 32)
      acc = wmma_bf16(a_frag_lds(sDb, SB_STR, k0),
                      b_frag_glb(Wp1T, DETER, n0, k0), acc);
    int col = n0 + colL;
    float bias = bp1[col];
#pragma unroll
    for (int v = 0; v < 8; ++v)
      sXb[(v + hi * 8) * SB_STR + col] = (bf16)elu_f(acc[v] + bias);
  }
  __syncthreads();

  // ---- Phase E: prior head (N=128, one n-tile per wave) -------------------
  {
    int n0 = wave * 16;
    v8f acc = {};
#pragma unroll 4
    for (int k0 = 0; k0 < MLP; k0 += 32)
      acc = wmma_bf16(a_frag_lds(sXb, SB_STR, k0),
                      b_frag_glb(Wp2T, MLP, n0, k0), acc);
    int col = n0 + colL;
    float bias = bp2[col];
#pragma unroll
    for (int v = 0; v < 8; ++v)
      sPR[(v + hi * 8) * SPR_STR + col] = acc[v] + bias;
  }
  __syncthreads();
  for (int idx = tid; idx < 16 * STOC; idx += 256) {
    int r = idx >> 6, c = idx & (STOC - 1);
    int b = m0 + r;
    float pm  = sPR[r * SPR_STR + c];
    float ps  = softplus_f(sPR[r * SPR_STR + STOC + c]) + 0.1f;
    float eps = noise_p[((size_t)t * BATCH + b) * STOC + c];
    float* orow = curOut + (size_t)b * OUTW;
    orow[DETER + c]            = pm;
    orow[DETER + STOC + c]     = ps;
    orow[DETER + 2 * STOC + c] = pm + ps * eps;
  }
  __syncthreads();

  // ---- Phase F: posterior hidden = elu([sD | emb] @ Wq1T + bq1) -> sXb ----
  for (int j = wave; j < MLP / 16; j += 8) {
    int n0 = j * 16;
    v8f acc = {};
#pragma unroll 4
    for (int k0 = 0; k0 < DETER; k0 += 32)
      acc = wmma_bf16(a_frag_lds(sDb, SB_STR, k0),
                      b_frag_glb(Wq1T, DETER + EMB, n0, k0), acc);
#pragma unroll 4
    for (int k0 = 0; k0 < EMB; k0 += 32)
      acc = wmma_bf16(a_frag_lds(sEmb, SEMB_STR, k0),
                      b_frag_glb(Wq1T, DETER + EMB, n0, DETER + k0), acc);
    int col = n0 + colL;
    float bias = bq1[col];
#pragma unroll
    for (int v = 0; v < 8; ++v)
      sXb[(v + hi * 8) * SB_STR + col] = (bf16)elu_f(acc[v] + bias);
  }
  __syncthreads();

  // ---- Phase G: posterior head -------------------------------------------
  {
    int n0 = wave * 16;
    v8f acc = {};
#pragma unroll 4
    for (int k0 = 0; k0 < MLP; k0 += 32)
      acc = wmma_bf16(a_frag_lds(sXb, SB_STR, k0),
                      b_frag_glb(Wq2T, MLP, n0, k0), acc);
    int col = n0 + colL;
    float bias = bq2[col];
#pragma unroll
    for (int v = 0; v < 8; ++v)
      sPR[(v + hi * 8) * SPR_STR + col] = acc[v] + bias;
  }
  __syncthreads();
  for (int idx = tid; idx < 16 * STOC; idx += 256) {
    int r = idx >> 6, c = idx & (STOC - 1);
    int b = m0 + r;
    float qm  = sPR[r * SPR_STR + c];
    float qs  = softplus_f(sPR[r * SPR_STR + STOC + c]) + 0.1f;
    float eps = noise_q[((size_t)t * BATCH + b) * STOC + c];
    float* orow = curOut + (size_t)b * OUTW;
    orow[DETER + 3 * STOC + c] = qm;
    orow[DETER + 4 * STOC + c] = qs;
    orow[DETER + 5 * STOC + c] = qm + qs * eps;  // qstoc = next stoc carry
  }
}

// ---------------------------------------------------------------------------
extern "C" void kernel_launch(void* const* d_in, const int* in_sizes, int n_in,
                              void* d_out, int out_size, void* d_ws,
                              size_t ws_size, hipStream_t stream) {
  const float* actions  = (const float*)d_in[0];
  const float* nonterm  = (const float*)d_in[1];
  const float* emb      = (const float*)d_in[2];
  const float* init_det = (const float*)d_in[3];
  const float* init_sto = (const float*)d_in[4];
  const float* noise_p  = (const float*)d_in[5];
  const float* noise_q  = (const float*)d_in[6];
  const float* Wi       = (const float*)d_in[7];
  const float* bi       = (const float*)d_in[8];
  const float* W_ih     = (const float*)d_in[9];
  const float* W_hh     = (const float*)d_in[10];
  const float* b_ih     = (const float*)d_in[11];
  const float* b_hh     = (const float*)d_in[12];
  const float* Wp1      = (const float*)d_in[13];
  const float* bp1      = (const float*)d_in[14];
  const float* Wp2      = (const float*)d_in[15];
  const float* bp2      = (const float*)d_in[16];
  const float* Wq1      = (const float*)d_in[17];
  const float* bq1      = (const float*)d_in[18];
  const float* Wq2      = (const float*)d_in[19];
  const float* bq2      = (const float*)d_in[20];

  char* ws = (char*)d_ws;
  size_t off = 0;
  auto alloc = [&](size_t bytes) -> bf16* {
    bf16* p = (bf16*)(ws + off);
    off += (bytes + 255) & ~(size_t)255;
    return p;
  };
  bf16* WiT  = alloc((size_t)DETER * KIN * 2);
  bf16* Wih  = alloc((size_t)3 * DETER * DETER * 2);
  bf16* Whh  = alloc((size_t)3 * DETER * DETER * 2);
  bf16* Wp1T = alloc((size_t)MLP * DETER * 2);
  bf16* Wp2T = alloc((size_t)2 * STOC * MLP * 2);
  bf16* Wq1T = alloc((size_t)MLP * (DETER + EMB) * 2);
  bf16* Wq2T = alloc((size_t)2 * STOC * MLP * 2);

  const int thr = 256;
  auto blocks = [](long n, int t) { return (int)((n + t - 1) / t); };

  // One-time (per call) weight conversion into L2-resident bf16 layouts.
  cvt_kernel<<<blocks(3L * DETER * DETER, thr), thr, 0, stream>>>(W_ih, Wih,
                                                                 3 * DETER * DETER);
  cvt_kernel<<<blocks(3L * DETER * DETER, thr), thr, 0, stream>>>(W_hh, Whh,
                                                                 3 * DETER * DETER);
  transpose_cvt_kernel<<<blocks((long)DETER * MLP, thr), thr, 0, stream>>>(
      Wp1, Wp1T, DETER, MLP);
  transpose_cvt_kernel<<<blocks((long)MLP * 2 * STOC, thr), thr, 0, stream>>>(
      Wp2, Wp2T, MLP, 2 * STOC);
  transpose_cvt_kernel<<<blocks((long)(DETER + EMB) * MLP, thr), thr, 0, stream>>>(
      Wq1, Wq1T, DETER + EMB, MLP);
  transpose_cvt_kernel<<<blocks((long)MLP * 2 * STOC, thr), thr, 0, stream>>>(
      Wq2, Wq2T, MLP, 2 * STOC);
  wi_pad_kernel<<<blocks((long)DETER * KIN, thr), thr, 0, stream>>>(Wi, WiT);

  float* out = (float*)d_out;
  for (int t = 0; t < T_STEPS; ++t) {
    rssm_step<<<BATCH / 16, 256, 0, stream>>>(
        t, actions, nonterm, emb, init_det, init_sto, noise_p, noise_q, bi,
        b_ih, b_hh, bp1, bp2, bq1, bq2, WiT, Wih, Whh, Wp1T, Wp2T, Wq1T, Wq2T,
        out);
  }
}
